// AdaConv_7756710936814
// MI455X (gfx1250) — matrix operand order repaired and checked by reading, not compiled, not run on hardware
//
#include <hip/hip_runtime.h>

typedef __bf16 bf16_t;
typedef __attribute__((ext_vector_type(2)))  __bf16 v2bf;
typedef __attribute__((ext_vector_type(16))) __bf16 v16bf;
typedef __attribute__((ext_vector_type(8)))  float  v8f;

// ---------------------------------------------------------------------------
// Prediction network (tiny, VALU): dw1 (1x1 conv + lrelu) on style [8,512,4,4]
// ---------------------------------------------------------------------------
__global__ void k_dw1(const float* __restrict__ style, const float* __restrict__ w,
                      const float* __restrict__ b, float* __restrict__ h) {
    int idx = blockIdx.x * blockDim.x + threadIdx.x;   // 8*512*16
    int s  = idx & 15;
    int co = (idx >> 4) & 511;
    int n  = idx >> 13;
    const float* xin = style + (size_t)n * 512 * 16 + s;
    const float* wr  = w + (size_t)co * 512;
    float acc = b[co];
#pragma unroll 8
    for (int ci = 0; ci < 512; ++ci) acc += wr[ci] * xin[ci * 16];
    h[idx] = acc > 0.f ? acc : 0.01f * acc;
}

// dw2: 2x2 VALID conv, h[8,512,4,4] -> dwk[8,2048,3,3] (no activation)
__global__ void k_dw2(const float* __restrict__ h, const float* __restrict__ w,
                      const float* __restrict__ b, float* __restrict__ dwk) {
    int idx = blockIdx.x * blockDim.x + threadIdx.x;   // 8*2048*9
    int p  = idx % 9;
    int oc = (idx / 9) & 2047;
    int n  = idx / (9 * 2048);
    int py = p / 3, px = p % 3;
    const float* hb = h + (size_t)n * 512 * 16;
    const float* wr = w + (size_t)oc * 512 * 4;
    float acc = b[oc];
#pragma unroll 4
    for (int ci = 0; ci < 512; ++ci) {
        const float* hc = hb + ci * 16;
        const float* wc = wr + ci * 4;
        acc += wc[0] * hc[py * 4 + px]       + wc[1] * hc[py * 4 + px + 1]
             + wc[2] * hc[(py + 1) * 4 + px] + wc[3] * hc[(py + 1) * 4 + px + 1];
    }
    dwk[idx] = acc;
}

// mean-pool style over 4x4 -> spool[8,512]
__global__ void k_pool(const float* __restrict__ style, float* __restrict__ spool) {
    int idx = blockIdx.x * blockDim.x + threadIdx.x;   // 4096
    const float* sp = style + (size_t)idx * 16;
    float acc = 0.f;
#pragma unroll
    for (int s = 0; s < 16; ++s) acc += sp[s];
    spool[idx] = acc * (1.f / 16.f);
}

// generic dense: out[i][o] = act(b[o] + W[o,:]·in[i,:])
__global__ void k_dense(const float* __restrict__ in, const float* __restrict__ w,
                        const float* __restrict__ b, float* __restrict__ out,
                        int K, int nOut, int act, int total) {
    int idx = blockIdx.x * blockDim.x + threadIdx.x;
    if (idx >= total) return;
    int o = idx % nOut;
    int n = idx / nOut;
    const float* wr = w + (size_t)o * K;
    const float* xr = in + (size_t)n * K;
    float acc = b[o];
#pragma unroll 8
    for (int k = 0; k < K; ++k) acc += wr[k] * xr[k];
    if (act) acc = acc > 0.f ? acc : 0.01f * acc;
    out[idx] = acc;
}

// ---------------------------------------------------------------------------
// Build the block-diagonal effective kernel directly in the per-lane WMMA
// A-fragment order:  keff[n][cb][t(0..4)][lane(0..31)][e(0..15)]  (bf16)
// Element (t,lane,e) holds Keff value for logical K = tap_local*16 + ci,
// with lane half h holding k = 16h + e (contiguous order, as the B side uses), so the
// fused kernel loads each fragment as one contiguous 32-byte chunk.
// ---------------------------------------------------------------------------
__global__ void k_keff(const float* __restrict__ pkw, const float* __restrict__ dwk,
                       bf16_t* __restrict__ keff) {
    int idx = blockIdx.x * blockDim.x + threadIdx.x;   // 8*32*5*32*16
    int e    = idx & 15;
    int lane = (idx >> 4) & 31;
    int t    = (idx >> 9) % 5;
    int cbn  = idx / 2560;          // n*32 + cb
    int cb   = cbn & 31;
    int n    = cbn >> 5;

    int hi  = lane >> 4;
    int row = lane & 15;            // out channel within 16-block (M)
    int v   = e >> 1, hh = e & 1;
    int k   = hi * 16 + e;
    int tl  = k >> 4;               // tap_local
    int ci  = k & 15;               // in channel within 16-block
    int tap = 2 * t + tl;

    float val = 0.f;
    if (tap < 9 && (row >> 2) == (ci >> 2)) {
        int g  = cb * 4 + (row >> 2);
        int ch = cb * 16 + row;
        int il = ci & 3;
        const float* kp = pkw + (size_t)n * 2048 + (size_t)ch * 4;
#pragma unroll
        for (int m = 0; m < 4; ++m)
            val += kp[m] * dwk[((size_t)n * 2048 + (size_t)(g * 4 + m) * 4 + il) * 9 + tap];
    }
    keff[idx] = (bf16_t)val;
}

// ---------------------------------------------------------------------------
// Fused apply: one pass over predicted [8,512,64,64] with reflect padding.
// LDS layout is K-major: xs[row][col][16ch] so each lane's B fragment is one
// contiguous 32B chunk -> 2x ds_load_b128 per fragment. Row 18 is kept zero:
// the out-of-range tap of the 5th WMMA reads it instead of cndmask'ing data.
// ---------------------------------------------------------------------------
__global__ __launch_bounds__(256) void k_fused(const float* __restrict__ xin,
                                               const bf16_t* __restrict__ keff,
                                               const float* __restrict__ bias,
                                               float* __restrict__ out) {
    __shared__ __align__(32) bf16_t xs[19][68][16];   // rows 0..17 data, row 18 zeros

    int bid   = blockIdx.x;
    int tileY = bid & 3;
    int cb    = (bid >> 2) & 31;
    int n     = bid >> 7;
    int tid   = threadIdx.x;

    // Stage reflect-padded tile: gx-fastest for coalesced global reads,
    // two channels per thread packed into one 32-bit LDS store.
    const float* xbase = xin + (size_t)(n * 512 + cb * 16) * 4096;
    for (int i = tid; i < 18 * 8 * 66; i += 256) {
        int c      = i % 66;
        int tmp    = i / 66;
        int cipair = tmp & 7;
        int r      = tmp >> 3;
        int ci0    = cipair * 2;
        int gy = tileY * 16 - 1 + r;
        gy = gy < 0 ? -gy : (gy > 63 ? 126 - gy : gy);
        int gx = c - 1;
        gx = gx < 0 ? -gx : (gx > 63 ? 126 - gx : gx);
        const float* p = xbase + (size_t)ci0 * 4096 + gy * 64 + gx;
        v2bf pk;
        pk.x = (bf16_t)p[0];
        pk.y = (bf16_t)p[4096];
        *(v2bf*)&xs[r][c][ci0] = pk;
    }
    // zero row 18 (source of the zero K-half of the 5th WMMA)
    {
        bf16_t* zr = &xs[18][0][0];
        const v2bf z2 = {};
        for (int i = tid; i < 68 * 8; i += 256) *(v2bf*)(zr + i * 2) = z2;
    }
    __syncthreads();

    int lane = tid & 31;
    int wave = tid >> 5;
    int colN = lane & 15;       // N (pixel col) for B/D
    int hi   = lane >> 4;       // lane half: K sub-range / D row block

    // A fragments: contiguous 32B per lane (pre-swizzled by k_keff).
    v16bf A[5];
    const bf16_t* kb = keff + (size_t)(n * 32 + cb) * 5 * 512;
#pragma unroll
    for (int t = 0; t < 5; ++t)
        A[t] = *(const v16bf*)(kb + (size_t)t * 512 + lane * 16);

    // bias folded into accumulator init (C operand of WMMA)
    float bv[8];
#pragma unroll
    for (int r = 0; r < 8; ++r) bv[r] = bias[n * 512 + cb * 16 + (r + 8 * hi)];

    // 64 D-tiles of 16x16 per block; 8 waves x 8 tiles; 5 WMMAs per tile.
    for (int tile = wave; tile < 64; tile += 8) {
        int py = tile >> 2;       // row within 16-row band
        int tx = tile & 3;        // 16-col tile
        v8f acc;
#pragma unroll
        for (int r = 0; r < 8; ++r) acc[r] = bv[r];
#pragma unroll
        for (int t = 0; t < 5; ++t) {
            int tap = 2 * t + hi;           // this lane-half supplies this tap
            bool ok = tap < 9;              // compile-time true for t<4
            int tapc = ok ? tap : 0;
            int dy   = tapc / 3 - 1, dx = tapc % 3 - 1;
            int rr   = ok ? (py + 1 + dy) : 18;            // zero row for tap 9
            int cc   = ok ? (1 + tx * 16 + colN + dx) : colN;
            v16bf B = *(const v16bf*)&xs[rr][cc][0];
            acc = __builtin_amdgcn_wmma_f32_16x16x32_bf16(
                false, A[t], false, B, (short)0, acc, false, false);
        }
        int y = tileY * 16 + py;
        int x = tx * 16 + colN;
        float* ob = out + ((size_t)(n * 512 + cb * 16 + 8 * hi) * 64 + y) * 64 + x;
#pragma unroll
        for (int r = 0; r < 8; ++r) ob[(size_t)r * 4096] = acc[r];
    }
}

// ---------------------------------------------------------------------------
extern "C" void kernel_launch(void* const* d_in, const int* in_sizes, int n_in,
                              void* d_out, int out_size, void* d_ws, size_t ws_size,
                              hipStream_t stream) {
    const float* style = (const float*)d_in[0];   // [8,512,4,4]
    const float* pred  = (const float*)d_in[1];   // [8,512,64,64]
    const float* dw1_w = (const float*)d_in[2];
    const float* dw1_b = (const float*)d_in[3];
    const float* dw2_w = (const float*)d_in[4];
    const float* dw2_b = (const float*)d_in[5];
    const float* pk1_w = (const float*)d_in[6];
    const float* pk1_b = (const float*)d_in[7];
    const float* pk2_w = (const float*)d_in[8];
    const float* pk2_b = (const float*)d_in[9];
    const float* pb1_w = (const float*)d_in[10];
    const float* pb1_b = (const float*)d_in[11];
    const float* pb2_w = (const float*)d_in[12];
    const float* pb2_b = (const float*)d_in[13];
    float* out = (float*)d_out;

    // workspace layout (bytes)
    char* ws = (char*)d_ws;
    float*  h     = (float*)(ws);                      // 8*512*16   (262144 B)
    float*  dwk   = (float*)(ws + 262144);             // 8*2048*9   (589824 B)
    float*  spool = (float*)(ws + 851968);             // 8*512
    float*  pkh   = spool + 4096;                      // 8*512
    float*  pkw   = pkh + 4096;                        // 8*2048
    float*  pbh   = pkw + 16384;                       // 8*512
    float*  pbias = pbh + 4096;                        // 8*512
    bf16_t* keff  = (bf16_t*)(pbias + 4096);           // 8*32*5*32*16 bf16 (1310720 B)

    k_dw1 <<<65536 / 256, 256, 0, stream>>>(style, dw1_w, dw1_b, h);
    k_dw2 <<<147456 / 256, 256, 0, stream>>>(h, dw2_w, dw2_b, dwk);
    k_pool<<<4096 / 256, 256, 0, stream>>>(style, spool);
    k_dense<<<4096 / 256, 256, 0, stream>>>(spool, pk1_w, pk1_b, pkh, 512, 512, 1, 4096);
    k_dense<<<16384 / 256, 256, 0, stream>>>(pkh, pk2_w, pk2_b, pkw, 512, 2048, 0, 16384);
    k_dense<<<4096 / 256, 256, 0, stream>>>(spool, pb1_w, pb1_b, pbh, 512, 512, 1, 4096);
    k_dense<<<4096 / 256, 256, 0, stream>>>(pbh, pb2_w, pb2_b, pbias, 512, 512, 0, 4096);
    k_keff<<<655360 / 256, 256, 0, stream>>>(pkw, dwk, keff);
    k_fused<<<1024, 256, 0, stream>>>(pred, keff, pbias, out);
}
